// MambaBlock_43679817401115
// MI455X (gfx1250) — matrix operand order repaired and checked by reading, not compiled, not run on hardware
//
#include <hip/hip_runtime.h>
#include <hip/hip_bf16.h>

// ---------------- problem constants (from the reference) ----------------
#define BD      2
#define LEN     2048
#define DMODEL  768
#define DINNER  1536
#define DSTATE  16
#define DCONV   4
#define DTRANK  96
#define NXCOLS  (DTRANK + 2 * DSTATE)   // 128
#define MROWS   (BD * LEN)              // 4096 rows (B*L)
#define LN_EPS  1e-5f

// scan chunking: serial chain 2048 -> 128 (+16-step fixup)
#define NCHUNK  16
#define CHLEN   (LEN / NCHUNK)          // 128

// LDS B-tile: 32 (K) x 64 (N), padded row stride 72 f16 (144B) -> conflict-free
#define BT_STRIDE 72

typedef __attribute__((ext_vector_type(16))) _Float16 v16h;
typedef __attribute__((ext_vector_type(8)))  _Float16 v8h;
typedef __attribute__((ext_vector_type(8)))  float    v8f;

// ======================================================================
// CDNA5 async global->LDS staging (CDNA5 ISA)
// ======================================================================
__device__ __forceinline__ void async_b128_to_lds(unsigned lds_byte_off,
                                                  const void* gaddr) {
    asm volatile("global_load_async_to_lds_b128 %0, %1, off"
                 :: "v"(lds_byte_off),
                    "v"((unsigned long long)(size_t)gaddr)
                 : "memory");
}
__device__ __forceinline__ void wait_async_le1() {
    asm volatile("s_wait_asynccnt 0x1" ::: "memory");
}
__device__ __forceinline__ void wait_async_le0() {
    asm volatile("s_wait_asynccnt 0x0" ::: "memory");
}

__device__ __forceinline__ v16h combine16(v8h lo, v8h hi) {
    v16h r;
#pragma unroll
    for (int i = 0; i < 8; ++i) { r[i] = lo[i]; r[8 + i] = hi[i]; }
    return r;
}

__device__ __forceinline__ float softplusf(float x) {
    return (x > 20.f) ? x : log1pf(__expf(x));
}

// ======================================================================
// f32 -> f16 elementwise conversion (weights)
// ======================================================================
__global__ void convert_f32_to_f16(const float* __restrict__ in,
                                   _Float16* __restrict__ out, int n) {
    int i = blockIdx.x * blockDim.x + threadIdx.x;
    if (i < n) out[i] = (_Float16)in[i];
}

// ======================================================================
// LayerNorm: one block per row, writes f16 normalized row
// ======================================================================
__global__ void __launch_bounds__(256)
ln_kernel(const float* __restrict__ x, const float* __restrict__ gam,
          const float* __restrict__ bet, _Float16* __restrict__ xn16) {
    __shared__ float s1[256], s2[256];
    const int row = blockIdx.x;
    const int tid = threadIdx.x;
    const float* xr = x + (size_t)row * DMODEL;
    float a = 0.f, b = 0.f;
    for (int i = tid; i < DMODEL; i += 256) { float v = xr[i]; a += v; b += v * v; }
    s1[tid] = a; s2[tid] = b;
    __syncthreads();
    for (int s = 128; s > 0; s >>= 1) {
        if (tid < s) { s1[tid] += s1[tid + s]; s2[tid] += s2[tid + s]; }
        __syncthreads();
    }
    const float mean = s1[0] * (1.f / DMODEL);
    const float var  = s2[0] * (1.f / DMODEL) - mean * mean;
    const float inv  = rsqrtf(var + LN_EPS);
    for (int i = tid; i < DMODEL; i += 256)
        xn16[(size_t)row * DMODEL + i] = (_Float16)((xr[i] - mean) * inv * gam[i] + bet[i]);
}

// ======================================================================
// WMMA GEMM with async-LDS double-buffered B staging.
//   C(MxN,f32) = A(MxK,f16) @ B(KxN,f16) [+ epilogue]
//   mode 0: plain store              (aux unused)
//   mode 1: softplus(acc + aux[n])   (aux = bias over N)
//   mode 2: acc + aux[m*N+n]         (aux = residual MxN)
// out16 (optional): f16 copy of the result.
// Block = 8 waves -> 128(M) x 64(N) tile; each wave one 16x64 strip.
// B tile (32x64 per K-step) staged once per block via
// global_load_async_to_lds_b128, consumed with per-element 16-bit LDS reads.
// Requires: M%128==0, N%64==0, K%32==0, lda%8==0, N%8==0.
// ======================================================================
__global__ void __launch_bounds__(256)
wmma_gemm(const _Float16* __restrict__ A, const _Float16* __restrict__ B,
          float* __restrict__ C, _Float16* __restrict__ out16,
          const float* __restrict__ aux,
          int M, int N, int K, int lda, int mode) {
    __shared__ __align__(16) _Float16 btile[2][32 * BT_STRIDE];

    const int tid     = threadIdx.x;
    const int lane    = tid & 31;
    const int wave    = tid >> 5;
    const int rowBase = blockIdx.y * 128 + wave * 16;
    const int nBase0  = blockIdx.x * 64;

    // --- staging map: thread tid covers tile row (tid>>3), 8 f16 at col (tid&7)*8
    const int srow = tid >> 3;
    const int scol = (tid & 7) * 8;
    const _Float16* gsrc = B + (size_t)srow * N + nBase0 + scol;
    const unsigned ldsoff0 =
        (unsigned)(unsigned long long)(const void*)&btile[0][srow * BT_STRIDE + scol];
    const unsigned ldsoff1 =
        (unsigned)(unsigned long long)(const void*)&btile[1][srow * BT_STRIDE + scol];

    // prologue: stage K-chunk 0 into buffer 0
    async_b128_to_lds(ldsoff0, gsrc);
    gsrc += (size_t)32 * N;

    // per-lane A pointer (A layout: lane m = lane%16; lanes>=16 offset +8 in K)
    const _Float16* pa =
        A + (size_t)(rowBase + (lane & 15)) * lda + ((lane >> 4) << 3);

    v8f acc[4];
#pragma unroll
    for (int t = 0; t < 4; ++t) acc[t] = (v8f)0.f;

    int phase = 0;
    for (int kk = 0; kk < K; kk += 32) {
        if (kk + 32 < K) {
            // stream next B tile into the other buffer while computing
            async_b128_to_lds(phase ? ldsoff0 : ldsoff1, gsrc);
            gsrc += (size_t)32 * N;
            wait_async_le1();                 // current buffer staged (in-order)
        } else {
            wait_async_le0();
        }
        __syncthreads();                      // publish staged tile to all waves

        __builtin_prefetch(pa + 64, 0, 3);    // global_prefetch_b8 on A stream
        v16h a = combine16(*(const v8h*)pa, *(const v8h*)(pa + 16));
        pa += 32;

        const _Float16* bt = &btile[phase][((lane >> 4) << 3) * BT_STRIDE + (lane & 15)];
#pragma unroll
        for (int t = 0; t < 4; ++t) {
            v16h b;
#pragma unroll
            for (int i = 0; i < 8; ++i) {
                b[i]     = bt[i * BT_STRIDE + 16 * t];
                b[8 + i] = bt[(16 + i) * BT_STRIDE + 16 * t];
            }
            acc[t] = __builtin_amdgcn_wmma_f32_16x16x32_f16(
                false, a, false, b, (short)0, acc[t], false, false);
        }
        __syncthreads();                      // tile consumed; safe to overwrite
        phase ^= 1;
    }

    // C/D layout: VGPR r -> M = r (+8 for lanes 16..31), N = lane%16
    const int n_lo   = lane & 15;
    const int m_half = (lane >> 4) << 3;
#pragma unroll
    for (int t = 0; t < 4; ++t) {
        const int n = nBase0 + 16 * t + n_lo;
#pragma unroll
        for (int r = 0; r < 8; ++r) {
            const int m = rowBase + m_half + r;
            float val = acc[t][r];
            if (mode == 1)      val = softplusf(val + aux[n]);
            else if (mode == 2) val += aux[(size_t)m * N + n];
            C[(size_t)m * N + n] = val;
            if (out16) out16[(size_t)m * N + n] = (_Float16)val;
        }
    }
}

// ======================================================================
// Depthwise causal conv (k=4) + bias + SiLU. One thread per (b,l,d).
// ======================================================================
__global__ void __launch_bounds__(256)
conv_silu_kernel(const float* __restrict__ uv, const float* __restrict__ cw,
                 const float* __restrict__ cb, float* __restrict__ uc32,
                 _Float16* __restrict__ uc16) {
    const int idx = blockIdx.x * blockDim.x + threadIdx.x;
    const int d = idx % DINNER;
    const int m = idx / DINNER;
    const int l = m % LEN;
    const int b = m / LEN;
    float acc = cb[d];
#pragma unroll
    for (int j = 0; j < DCONV; ++j) {
        const int lj = l - (DCONV - 1) + j;
        if (lj >= 0)
            acc += cw[d * DCONV + j] * uv[((size_t)b * LEN + lj) * (2 * DINNER) + d];
    }
    const float s = acc / (1.f + __expf(-acc));              // SiLU
    uc32[idx] = s;
    uc16[idx] = (_Float16)s;
}

// ======================================================================
// Chunked selective scan.
// Phase 1: per chunk (h_in = 0) accumulate h_end[n] and aprod[n] = prod(a_bar).
// Fixup  : per (b,d,n) serial 16-step chain -> h_in per chunk.
// Phase 3: replay chunk with correct h_in; fuse u*D, SiLU gate, f16 store.
// Chunk-state layout: [b][c][n][d] (d fastest -> coalesced).
// ======================================================================
__global__ void __launch_bounds__(256)
scan_phase1(const float* __restrict__ delta, const float* __restrict__ xdbl,
            const float* __restrict__ uc,    const float* __restrict__ logA,
            float* __restrict__ aprod, float* __restrict__ hend) {
    __shared__ float sB[DSTATE];
    const int d  = blockIdx.x * blockDim.x + threadIdx.x;   // < DINNER
    const int bc = blockIdx.y;                              // b*NCHUNK + c
    const int b  = bc / NCHUNK;
    const int c  = bc % NCHUNK;

    float Arow[DSTATE], h[DSTATE], ap[DSTATE];
#pragma unroll
    for (int n = 0; n < DSTATE; ++n) {
        Arow[n] = -__expf(logA[(size_t)d * DSTATE + n]);
        h[n] = 0.f; ap[n] = 1.f;
    }

    const int l0 = c * CHLEN;
    for (int l = l0; l < l0 + CHLEN; ++l) {
        const size_t row = (size_t)b * LEN + l;
        __syncthreads();
        if (threadIdx.x < DSTATE)
            sB[threadIdx.x] = xdbl[row * NXCOLS + DTRANK + threadIdx.x];
        __syncthreads();
        const float dl  = delta[row * DINNER + d];
        const float dbu = dl * uc[row * DINNER + d];
#pragma unroll
        for (int n = 0; n < DSTATE; ++n) {
            const float abar = __expf(dl * Arow[n]);
            h[n]  = abar * h[n] + dbu * sB[n];
            ap[n] *= abar;
        }
    }
#pragma unroll
    for (int n = 0; n < DSTATE; ++n) {
        const size_t i = (((size_t)bc) * DSTATE + n) * DINNER + d;
        aprod[i] = ap[n];
        hend[i]  = h[n];
    }
}

__global__ void __launch_bounds__(256)
scan_fix(const float* __restrict__ aprod, const float* __restrict__ hend,
         float* __restrict__ hin) {
    const int gid = blockIdx.x * blockDim.x + threadIdx.x;  // < BD*DSTATE*DINNER
    const int d   = gid % DINNER;
    const int rem = gid / DINNER;
    const int n   = rem % DSTATE;
    const int b   = rem / DSTATE;
    float h = 0.f;
    for (int c = 0; c < NCHUNK; ++c) {
        const size_t i = (((size_t)(b * NCHUNK + c)) * DSTATE + n) * DINNER + d;
        hin[i] = h;
        h = aprod[i] * h + hend[i];
    }
}

__global__ void __launch_bounds__(256)
scan_phase3(const float* __restrict__ delta, const float* __restrict__ xdbl,
            const float* __restrict__ uc,    const float* __restrict__ uv,
            const float* __restrict__ logA,  const float* __restrict__ Dp,
            const float* __restrict__ hin,   _Float16* __restrict__ g16) {
    __shared__ float sB[DSTATE], sC[DSTATE];
    const int d  = blockIdx.x * blockDim.x + threadIdx.x;
    const int bc = blockIdx.y;
    const int b  = bc / NCHUNK;
    const int c  = bc % NCHUNK;

    float Arow[DSTATE], h[DSTATE];
#pragma unroll
    for (int n = 0; n < DSTATE; ++n) {
        Arow[n] = -__expf(logA[(size_t)d * DSTATE + n]);
        h[n] = hin[(((size_t)bc) * DSTATE + n) * DINNER + d];
    }
    const float Dd = Dp[d];

    const int l0 = c * CHLEN;
    for (int l = l0; l < l0 + CHLEN; ++l) {
        const size_t row = (size_t)b * LEN + l;
        __syncthreads();
        if (threadIdx.x < 2 * DSTATE) {
            const float v = xdbl[row * NXCOLS + DTRANK + threadIdx.x];
            if (threadIdx.x < DSTATE) sB[threadIdx.x] = v;
            else                      sC[threadIdx.x - DSTATE] = v;
        }
        __syncthreads();

        const float dl   = delta[row * DINNER + d];
        const float uval = uc[row * DINNER + d];
        const float dbu  = dl * uval;
        float y = 0.f;
#pragma unroll
        for (int n = 0; n < DSTATE; ++n) {
            const float abar = __expf(dl * Arow[n]);
            h[n] = abar * h[n] + dbu * sB[n];
            y += h[n] * sC[n];
        }
        y += uval * Dd;
        const float vv   = uv[row * (2 * DINNER) + DINNER + d];
        const float gate = vv / (1.f + __expf(-vv));         // silu(v)
        g16[row * DINNER + d] = (_Float16)(y * gate);
    }
}

// ======================================================================
// Host launcher
// ======================================================================
extern "C" void kernel_launch(void* const* d_in, const int* in_sizes, int n_in,
                              void* d_out, int out_size, void* d_ws, size_t ws_size,
                              hipStream_t stream) {
    const float* x       = (const float*)d_in[0];
    const float* ln_g    = (const float*)d_in[1];
    const float* ln_b    = (const float*)d_in[2];
    const float* W_in    = (const float*)d_in[3];
    const float* conv_w  = (const float*)d_in[4];
    const float* conv_b  = (const float*)d_in[5];
    const float* W_xproj = (const float*)d_in[6];
    const float* W_dt    = (const float*)d_in[7];
    const float* b_dt    = (const float*)d_in[8];
    const float* log_A   = (const float*)d_in[9];
    const float* D_param = (const float*)d_in[10];
    const float* W_out   = (const float*)d_in[11];
    float* out = (float*)d_out;

    char* ws = (char*)d_ws;
    size_t off = 0;
    auto bump = [&](size_t bytes) -> void* {
        off = (off + 255) & ~(size_t)255;
        void* p = ws + off;
        off += bytes;
        return p;
    };

    _Float16* Win16  = (_Float16*)bump((size_t)DMODEL * 2 * DINNER * 2);
    _Float16* Wxp16  = (_Float16*)bump((size_t)DINNER * NXCOLS * 2);
    _Float16* Wdt16  = (_Float16*)bump((size_t)DTRANK * DINNER * 2);
    _Float16* Wout16 = (_Float16*)bump((size_t)DINNER * DMODEL * 2);
    _Float16* xn16   = (_Float16*)bump((size_t)MROWS * DMODEL * 2);
    float*    uvbuf  = (float*)   bump((size_t)MROWS * 2 * DINNER * 4);
    float*    uc32   = (float*)   bump((size_t)MROWS * DINNER * 4);
    _Float16* uc16   = (_Float16*)bump((size_t)MROWS * DINNER * 2);
    float*    xdbl   = (float*)   bump((size_t)MROWS * NXCOLS * 4);
    _Float16* xdbl16 = (_Float16*)bump((size_t)MROWS * NXCOLS * 2);
    float*    dlt    = (float*)   bump((size_t)MROWS * DINNER * 4);
    _Float16* g16    = (_Float16*)bump((size_t)MROWS * DINNER * 2);
    const size_t chunkStates = (size_t)BD * NCHUNK * DSTATE * DINNER;
    float*    aprod  = (float*)   bump(chunkStates * 4);
    float*    hend   = (float*)   bump(chunkStates * 4);
    float*    hin    = (float*)   bump(chunkStates * 4);
    (void)ws_size; (void)in_sizes; (void)n_in; (void)out_size;

    const dim3 blk(256, 1, 1);

    // --- weight conversions to f16 ---
    {
        int n;
        n = DMODEL * 2 * DINNER;
        convert_f32_to_f16<<<dim3((n + 255) / 256), blk, 0, stream>>>(W_in, Win16, n);
        n = DINNER * NXCOLS;
        convert_f32_to_f16<<<dim3((n + 255) / 256), blk, 0, stream>>>(W_xproj, Wxp16, n);
        n = DTRANK * DINNER;
        convert_f32_to_f16<<<dim3((n + 255) / 256), blk, 0, stream>>>(W_dt, Wdt16, n);
        n = DINNER * DMODEL;
        convert_f32_to_f16<<<dim3((n + 255) / 256), blk, 0, stream>>>(W_out, Wout16, n);
    }

    // --- LayerNorm -> xn16 ---
    ln_kernel<<<dim3(MROWS), blk, 0, stream>>>(x, ln_g, ln_b, xn16);

    // --- in-projection: uv = xn16 @ W_in16  (4096 x 3072, K=768) ---
    wmma_gemm<<<dim3((2 * DINNER) / 64, MROWS / 128), blk, 0, stream>>>(
        xn16, Win16, uvbuf, nullptr, nullptr, MROWS, 2 * DINNER, DMODEL, DMODEL, 0);

    // --- causal depthwise conv + SiLU -> uc32 / uc16 ---
    conv_silu_kernel<<<dim3((MROWS * DINNER) / 256), blk, 0, stream>>>(
        uvbuf, conv_w, conv_b, uc32, uc16);

    // --- x-projection: x_dbl = uc16 @ W_xproj16  (4096 x 128, K=1536) ---
    wmma_gemm<<<dim3(NXCOLS / 64, MROWS / 128), blk, 0, stream>>>(
        uc16, Wxp16, xdbl, xdbl16, nullptr, MROWS, NXCOLS, DINNER, DINNER, 0);

    // --- dt-projection + softplus: delta = softplus(dlt @ W_dt + b_dt)
    //     (4096 x 1536, K=96; A = first 96 cols of x_dbl16, lda=128) ---
    wmma_gemm<<<dim3(DINNER / 64, MROWS / 128), blk, 0, stream>>>(
        xdbl16, Wdt16, dlt, nullptr, b_dt, MROWS, DINNER, DTRANK, NXCOLS, 1);

    // --- chunked selective scan -> g16 ---
    scan_phase1<<<dim3(DINNER / 256, BD * NCHUNK), blk, 0, stream>>>(
        dlt, xdbl, uc32, log_A, aprod, hend);
    scan_fix<<<dim3((BD * DSTATE * DINNER) / 256), blk, 0, stream>>>(
        aprod, hend, hin);
    scan_phase3<<<dim3(DINNER / 256, BD * NCHUNK), blk, 0, stream>>>(
        dlt, xdbl, uc32, uvbuf, log_A, D_param, hin, g16);

    // --- out-projection + residual: out = g16 @ W_out16 + x  (4096 x 768, K=1536) ---
    wmma_gemm<<<dim3(DMODEL / 64, MROWS / 128), blk, 0, stream>>>(
        g16, Wout16, out, nullptr, x, MROWS, DMODEL, DINNER, DINNER, 2);
}
